// RetinaNet_44332652429985
// MI455X (gfx1250) — compile-verified, matrix operand order checked by reading, not run on hardware
//
#include <hip/hip_runtime.h>
#include <hip/hip_bf16.h>

typedef __attribute__((ext_vector_type(16))) _Float16 v16h;
typedef __attribute__((ext_vector_type(8)))  float    v8f;

#define BM 128
#define BN 128
#define BK 32
#define NTOT 49104   // 9*(64^2+32^2+16^2+8^2+4^2)
#define PRE 1000
#define POST 300

// CDNA5 async global->LDS DMA path (ASYNCcnt), guarded so both toolchains and
// the host pass compile. Fallback: global_load_b128 + ds_store_b128.
#if defined(__gfx1250__) && __has_builtin(__builtin_amdgcn_global_load_async_to_lds_b128) && \
    __has_builtin(__builtin_amdgcn_s_wait_asynccnt)
#define USE_ASYNC_LDS 1
#else
#define USE_ASYNC_LDS 0
#endif

#if USE_ASYNC_LDS
// builtin signature (from hipcc diagnostic): (int4vec AS1*, int4vec AS3*, imm int, imm int)
typedef int i32x4_ __attribute__((vector_size(16)));
#define AS1V4(p) ((__attribute__((address_space(1))) i32x4_*)(p))
#define AS3V4(p) ((__attribute__((address_space(3))) i32x4_*)(p))
#endif

// ---------------------------------------------------------------------------
// Implicit-GEMM convolution, f16 data, f32 accumulate, WMMA 16x16x32.
// Double-buffered LDS tiles; async DMA staging when available.
// act: [B,Hin,Win,Cin] NHWC f16
// wgt: [Cout][KH*KW*Cin] f16 with k = (kh*KW+kw)*Cin + cin
// out: [B,Hout,Wout,Cout] NHWC f16
// ---------------------------------------------------------------------------
__global__ __launch_bounds__(256) void conv_implicit_gemm_wmma(
    const _Float16* __restrict__ act, const _Float16* __restrict__ wgt,
    const float* __restrict__ bias, _Float16* __restrict__ out,
    int Bn, int Hin, int Win, int Cin, int Hout, int Wout, int Cout,
    int KH, int KW, int stride, int pad, int relu)
{
  const int M  = Bn * Hout * Wout;
  const int K  = KH * KW * Cin;
  const int HW = Hout * Wout;
  const int tid  = threadIdx.x;
  const int lane = tid & 31;
  const int wave = tid >> 5;
  const int wm = wave & 3;     // 4 waves along M, 32 rows each
  const int wn = wave >> 2;    // 2 waves along N, 64 cols each
  const int m0 = blockIdx.x * BM;
  const int n0 = blockIdx.y * BN;

  __shared__ _Float16 As[2][BM * BK];   // [m][k]
  __shared__ _Float16 Bs[2][BN * BK];   // [n][k]

  v8f acc[2][4] = {};

  const bool fastA = (Cin % 8) == 0;
  const bool fastB = (K % 8) == 0;

  // stage one K-slab (32 k-values) of A and B into LDS buffer `buf`
  auto stage = [&](int kt, int buf) {
    _Float16* Ab = As[buf];
    _Float16* Bb = Bs[buf];
    const uint4 zero4 = {0u, 0u, 0u, 0u};
    // ---- A tile ----
    if (fastA) {
      #pragma unroll
      for (int i = 0; i < 2; i++) {
        int vid  = tid * 2 + i;          // 0..511 16B vectors
        int mr   = vid >> 2;
        int kseg = (vid & 3) * 8;
        int m = m0 + mr;
        int k = kt + kseg;
        bool inb = false;
        const _Float16* src = act;
        if (m < M && k < K) {
          int b  = m / HW; int rem = m - b * HW;
          int oh = rem / Wout; int ow = rem - oh * Wout;
          int sp = k / Cin; int cin = k - sp * Cin;
          int kh = sp / KW; int kw = sp - kh * KW;
          int ih = oh * stride + kh - pad;
          int iw = ow * stride + kw - pad;
          if (ih >= 0 && ih < Hin && iw >= 0 && iw < Win) {
            src = act + ((((long)b * Hin + ih) * Win + iw) * Cin + cin);
            inb = true;
          }
        }
        _Float16* dst = &Ab[mr * BK + kseg];
        if (inb) {
#if USE_ASYNC_LDS
          __builtin_amdgcn_global_load_async_to_lds_b128(
              AS1V4(const_cast<_Float16*>(src)), AS3V4(dst), 0, 0);
#else
          *(uint4*)dst = *(const uint4*)src;
#endif
        } else {
          *(uint4*)dst = zero4;
        }
      }
    } else {
      #pragma unroll
      for (int j = 0; j < 16; j++) {
        int e  = tid + j * 256;          // 0..4095 scalars
        int mr = e >> 5; int kk = e & 31;
        int m = m0 + mr; int k = kt + kk;
        _Float16 v = (_Float16)0.0f;
        if (m < M && k < K) {
          int b  = m / HW; int rem = m - b * HW;
          int oh = rem / Wout; int ow = rem - oh * Wout;
          int sp = k / Cin; int cin = k - sp * Cin;
          int kh = sp / KW; int kw = sp - kh * KW;
          int ih = oh * stride + kh - pad;
          int iw = ow * stride + kw - pad;
          if (ih >= 0 && ih < Hin && iw >= 0 && iw < Win)
            v = act[(((long)b * Hin + ih) * Win + iw) * Cin + cin];
        }
        Ab[mr * BK + kk] = v;
      }
    }
    // ---- B tile ----
    if (fastB) {
      #pragma unroll
      for (int i = 0; i < 2; i++) {
        int vid  = tid * 2 + i;
        int nr   = vid >> 2;
        int kseg = (vid & 3) * 8;
        int n = n0 + nr; int k = kt + kseg;
        _Float16* dst = &Bb[nr * BK + kseg];
        if (n < Cout && k < K) {
          const _Float16* src = wgt + (long)n * K + k;
#if USE_ASYNC_LDS
          __builtin_amdgcn_global_load_async_to_lds_b128(
              AS1V4(const_cast<_Float16*>(src)), AS3V4(dst), 0, 0);
#else
          *(uint4*)dst = *(const uint4*)src;
#endif
        } else {
          *(uint4*)dst = zero4;
        }
      }
    } else {
      #pragma unroll
      for (int j = 0; j < 16; j++) {
        int e  = tid + j * 256;
        int nr = e >> 5; int kk = e & 31;
        int n = n0 + nr; int k = kt + kk;
        _Float16 v = (_Float16)0.0f;
        if (n < Cout && k < K) v = wgt[(long)n * K + k];
        Bb[nr * BK + kk] = v;
      }
    }
  };

  const int nk = (K + BK - 1) / BK;
  stage(0, 0);
  int cur = 0;

  for (int t = 0; t < nk; t++) {
#if USE_ASYNC_LDS
    __builtin_amdgcn_s_wait_asynccnt(0);   // own DMAs done; barrier makes it WG-wide
#endif
    __syncthreads();

    if (t + 1 < nk) {
      stage((t + 1) * BK, cur ^ 1);        // overlap next slab with this slab's WMMAs
#if !USE_ASYNC_LDS
      int nr = n0 + (tid & 127);
      if (nr < Cout) __builtin_prefetch((const void*)(wgt + (long)nr * K + (t + 1) * BK), 0, 1);
#endif
    }

    // ---- fragments + WMMA on buffer `cur` ----
    const _Float16* Ab = As[cur];
    const _Float16* Bb = Bs[cur];
    const int lh  = lane >> 4;
    const int l15 = lane & 15;
    union Frag { v16h v; uint4 q[2]; };
    Frag afr[2], bfr[4];
    #pragma unroll
    for (int mi = 0; mi < 2; mi++) {
      int row = wm * 32 + mi * 16 + l15;
      afr[mi].q[0] = *(const uint4*)(&Ab[row * BK + lh * 8]);
      afr[mi].q[1] = *(const uint4*)(&Ab[row * BK + 16 + lh * 8]);
    }
    #pragma unroll
    for (int ni = 0; ni < 4; ni++) {
      int col = wn * 64 + ni * 16 + l15;
      bfr[ni].q[0] = *(const uint4*)(&Bb[col * BK + lh * 8]);
      bfr[ni].q[1] = *(const uint4*)(&Bb[col * BK + 16 + lh * 8]);
    }
    #pragma unroll
    for (int mi = 0; mi < 2; mi++)
      #pragma unroll
      for (int ni = 0; ni < 4; ni++)
        acc[mi][ni] = __builtin_amdgcn_wmma_f32_16x16x32_f16(
            false, afr[mi].v, false, bfr[ni].v, (short)0, acc[mi][ni], false, false);

    __syncthreads();   // all waves done reading `cur` before it is refilled
    cur ^= 1;
  }

  // ---------------- epilogue: bias (+relu), store NHWC f16 ----------------
  const int lh  = lane >> 4;
  const int l15 = lane & 15;
  #pragma unroll
  for (int ni = 0; ni < 4; ni++) {
    int gcol = n0 + wn * 64 + ni * 16 + l15;
    float bv = (gcol < Cout) ? bias[gcol] : 0.0f;
    #pragma unroll
    for (int mi = 0; mi < 2; mi++) {
      #pragma unroll
      for (int r = 0; r < 8; r++) {
        int gm = m0 + wm * 32 + mi * 16 + lh * 8 + r;
        if (gm < M && gcol < Cout) {
          float v = acc[mi][ni][r] + bv;
          if (relu) v = fmaxf(v, 0.0f);
          out[(long)gm * Cout + gcol] = (_Float16)v;
        }
      }
    }
  }
}

// ---------------------------------------------------------------------------
// Elementwise / layout kernels
// ---------------------------------------------------------------------------
__global__ void k_nchw_to_nhwc_f16(const float* __restrict__ in, _Float16* __restrict__ out,
                                   int B, int C, int H, int W) {
  long total = (long)B * C * H * W;
  for (long i = (long)blockIdx.x * blockDim.x + threadIdx.x; i < total;
       i += (long)gridDim.x * blockDim.x) {
    long t = i; int c = (int)(t % C); t /= C; int w = (int)(t % W); t /= W;
    int h = (int)(t % H); int b = (int)(t / H);
    out[i] = (_Float16)in[(((long)b * C + c) * H + h) * W + w];
  }
}

// OIHW f32 -> [Cout][(kh*KW+kw)*Cin + cin] f16
__global__ void k_repack_w(const float* __restrict__ in, _Float16* __restrict__ out,
                           int Cout, int Cin, int KH, int KW) {
  long total = (long)Cout * Cin * KH * KW;
  for (long i = (long)blockIdx.x * blockDim.x + threadIdx.x; i < total;
       i += (long)gridDim.x * blockDim.x) {
    long t = i; int cin = (int)(t % Cin); t /= Cin; int kw = (int)(t % KW); t /= KW;
    int kh = (int)(t % KH); int o = (int)(t / KH);
    out[i] = (_Float16)in[((((long)o * Cin + cin) * KH + kh) * KW + kw)];
  }
}

__global__ void k_maxpool2(const _Float16* __restrict__ in, _Float16* __restrict__ out,
                           int B, int H, int W, int C) {
  int Ho = H / 2, Wo = W / 2;
  long total = (long)B * Ho * Wo * C;
  for (long i = (long)blockIdx.x * blockDim.x + threadIdx.x; i < total;
       i += (long)gridDim.x * blockDim.x) {
    long t = i; int c = (int)(t % C); t /= C; int wo = (int)(t % Wo); t /= Wo;
    int ho = (int)(t % Ho); int b = (int)(t / Ho);
    float m = -1e30f;
    #pragma unroll
    for (int dy = 0; dy < 2; dy++)
      #pragma unroll
      for (int dx = 0; dx < 2; dx++) {
        float v = (float)in[((((long)b * H + (ho * 2 + dy)) * W + (wo * 2 + dx)) * C + c)];
        m = fmaxf(m, v);
      }
    out[i] = (_Float16)m;
  }
}

// out[b,h,w,c] = a[b,h,w,c] + s[b,h/2,w/2,c]   (2x nearest upsample + add)
__global__ void k_add_up2(const _Float16* __restrict__ a, const _Float16* __restrict__ s,
                          _Float16* __restrict__ out, int B, int H, int W, int C) {
  int Hs = H / 2, Ws = W / 2;
  long total = (long)B * H * W * C;
  for (long i = (long)blockIdx.x * blockDim.x + threadIdx.x; i < total;
       i += (long)gridDim.x * blockDim.x) {
    long t = i; int c = (int)(t % C); t /= C; int w = (int)(t % W); t /= W;
    int h = (int)(t % H); int b = (int)(t / H);
    float v = (float)a[i] + (float)s[((((long)b * Hs + h / 2) * Ws + w / 2) * C + c)];
    out[i] = (_Float16)v;
  }
}

__global__ void k_relu_f16(const _Float16* __restrict__ in, _Float16* __restrict__ out, long n) {
  for (long i = (long)blockIdx.x * blockDim.x + threadIdx.x; i < n;
       i += (long)gridDim.x * blockDim.x) {
    float v = (float)in[i];
    out[i] = (_Float16)fmaxf(v, 0.0f);
  }
}

// cls raw f16 [B][lvlElems] -> sigmoid -> classify f32 [B][Ntot*90] chunk
__global__ void k_sigmoid_store(const _Float16* __restrict__ in, float* __restrict__ out,
                                int B, long lvlElems, long strideB, long offset) {
  long total = (long)B * lvlElems;
  for (long i = (long)blockIdx.x * blockDim.x + threadIdx.x; i < total;
       i += (long)gridDim.x * blockDim.x) {
    long b = i / lvlElems, j = i - b * lvlElems;
    float x = (float)in[i];
    out[b * strideB + offset + j] = 1.0f / (1.0f + expf(-x));
  }
}

// anchor gen + box decode for one pyramid level
__global__ void k_decode(const _Float16* __restrict__ reg, float* __restrict__ boxes,
                         int B, int H, int W, float stride_, int ntot, long baseAnchor) {
  long lvlN = (long)H * W * 9;
  long total = (long)B * lvlN;
  for (long i = (long)blockIdx.x * blockDim.x + threadIdx.x; i < total;
       i += (long)gridDim.x * blockDim.x) {
    long b = i / lvlN, j = i - b * lvlN;
    int a = (int)(j % 9); long loc = j / 9;
    int w = (int)(loc % W); int h = (int)(loc / W);
    int r = a / 3, si = a % 3;
    float scale = exp2f((float)si / 3.0f);
    float ratio = (r == 0) ? 0.5f : (r == 1 ? 1.0f : 2.0f);
    float base = 4.0f * stride_;
    float sq = sqrtf(ratio);
    float ws = base * scale / sq, hs = base * scale * sq;
    float cx = ((float)w + 0.5f) * stride_, cy = ((float)h + 0.5f) * stride_;
    float x0 = cx - 0.5f * ws, y0 = cy - 0.5f * hs;
    float x1 = cx + 0.5f * ws, y1 = cy + 0.5f * hs;
    float wa = x1 - x0, ha = y1 - y0;
    float cxa = x0 + 0.5f * wa, cya = y0 + 0.5f * ha;
    const _Float16* d = reg + i * 4;
    float dx = (float)d[0], dy = (float)d[1], dw = (float)d[2], dh = (float)d[3];
    float ncx = dx * wa + cxa, ncy = dy * ha + cya;
    float nw = expf(fminf(fmaxf(dw, -4.0f), 4.0f)) * wa;
    float nh = expf(fminf(fmaxf(dh, -4.0f), 4.0f)) * ha;
    float* o = boxes + ((long)b * ntot + baseAnchor + j) * 4;
    o[0] = fminf(fmaxf(ncx - 0.5f * nw, 0.0f), 512.0f);
    o[1] = fminf(fmaxf(ncy - 0.5f * nh, 0.0f), 512.0f);
    o[2] = fminf(fmaxf(ncx + 0.5f * nw, 0.0f), 512.0f);
    o[3] = fminf(fmaxf(ncy + 0.5f * nh, 0.0f), 512.0f);
  }
}

__global__ void k_maxclass(const float* __restrict__ cls, float* __restrict__ maxs,
                           int* __restrict__ lab, int B, int ntot) {
  long total = (long)B * ntot;
  for (long i = (long)blockIdx.x * blockDim.x + threadIdx.x; i < total;
       i += (long)gridDim.x * blockDim.x) {
    const float* p = cls + i * 90;
    float mv = p[0]; int mi = 0;
    for (int c = 1; c < 90; c++) { float v = p[c]; if (v > mv) { mv = v; mi = c; } }
    maxs[i] = (mv > 0.05f) ? mv : -1.0f;
    lab[i] = mi;
  }
}

// per-image: top-1000, greedy NMS, top-300 write-out
__global__ __launch_bounds__(256) void k_nms(const float* __restrict__ maxs,
                                             const int* __restrict__ lab,
                                             const float* __restrict__ boxes,
                                             float* __restrict__ swork,
                                             float* __restrict__ outB,
                                             float* __restrict__ outS,
                                             float* __restrict__ outL, int ntot) {
  int b = blockIdx.x;
  const float* S = maxs + (long)b * ntot;
  const int* L = lab + (long)b * ntot;
  const float* BX = boxes + (long)b * ntot * 4;
  float* WK = swork + (long)b * ntot;
  int tid = threadIdx.x;

  __shared__ float rv[256]; __shared__ int ri[256];
  __shared__ float tb[PRE * 4];
  __shared__ float ts[PRE];
  __shared__ int tl_[PRE];
  __shared__ int keepf[PRE];
  __shared__ float fs[PRE];

  for (int i = tid; i < ntot; i += 256) WK[i] = S[i];
  __syncthreads();

  // top-PRE by iterative block argmax
  for (int sel = 0; sel < PRE; sel++) {
    float bv = -1e30f; int bi = 0;
    for (int i = tid; i < ntot; i += 256) { float v = WK[i]; if (v > bv) { bv = v; bi = i; } }
    rv[tid] = bv; ri[tid] = bi; __syncthreads();
    for (int off = 128; off > 0; off >>= 1) {
      if (tid < off && rv[tid + off] > rv[tid]) { rv[tid] = rv[tid + off]; ri[tid] = ri[tid + off]; }
      __syncthreads();
    }
    if (tid == 0) {
      int idx = ri[0];
      ts[sel] = rv[0];
      tl_[sel] = L[idx];
      tb[sel * 4 + 0] = BX[idx * 4 + 0];
      tb[sel * 4 + 1] = BX[idx * 4 + 1];
      tb[sel * 4 + 2] = BX[idx * 4 + 2];
      tb[sel * 4 + 3] = BX[idx * 4 + 3];
      keepf[sel] = (rv[0] > 0.0f) ? 1 : 0;
      WK[idx] = -1e30f;
    }
    __syncthreads();
  }

  // greedy suppression
  for (int i = 0; i < PRE; i++) {
    if (keepf[i]) {
      float x0 = tb[i * 4], y0 = tb[i * 4 + 1], x1 = tb[i * 4 + 2], y1 = tb[i * 4 + 3];
      float ai = fmaxf(x1 - x0, 0.0f) * fmaxf(y1 - y0, 0.0f);
      for (int j = i + 1 + tid; j < PRE; j += 256) {
        if (!keepf[j]) continue;
        float u0 = tb[j * 4], v0 = tb[j * 4 + 1], u1 = tb[j * 4 + 2], v1 = tb[j * 4 + 3];
        float aj = fmaxf(u1 - u0, 0.0f) * fmaxf(v1 - v0, 0.0f);
        float ix = fminf(x1, u1) - fmaxf(x0, u0);
        float iy = fminf(y1, v1) - fmaxf(y0, v0);
        float inter = fmaxf(ix, 0.0f) * fmaxf(iy, 0.0f);
        float iou = inter / (ai + aj - inter + 1e-8f);
        if (iou > 0.5f) keepf[j] = 0;
      }
    }
    __syncthreads();
  }

  for (int i = tid; i < PRE; i += 256) fs[i] = keepf[i] ? ts[i] : -1.0f;
  __syncthreads();

  // top-POST write-out
  for (int sel = 0; sel < POST; sel++) {
    float bv = -1e30f; int bi = 0;
    for (int i = tid; i < PRE; i += 256) { if (fs[i] > bv) { bv = fs[i]; bi = i; } }
    rv[tid] = bv; ri[tid] = bi; __syncthreads();
    for (int off = 128; off > 0; off >>= 1) {
      if (tid < off && rv[tid + off] > rv[tid]) { rv[tid] = rv[tid + off]; ri[tid] = ri[tid + off]; }
      __syncthreads();
    }
    if (tid == 0) {
      int idx = ri[0]; float sv = rv[0];
      long o = (long)b * POST + sel;
      outB[o * 4 + 0] = tb[idx * 4 + 0];
      outB[o * 4 + 1] = tb[idx * 4 + 1];
      outB[o * 4 + 2] = tb[idx * 4 + 2];
      outB[o * 4 + 3] = tb[idx * 4 + 3];
      outS[o] = (sv > 0.0f) ? sv : 0.0f;
      outL[o] = (float)tl_[idx];
      fs[idx] = -1e30f;
    }
    __syncthreads();
  }
}

// ---------------------------------------------------------------------------
// Host-side orchestration
// ---------------------------------------------------------------------------
static inline int eg(long n) { long b = (n + 255) / 256; return (int)(b > 65535 ? 65535 : (b < 1 ? 1 : b)); }

static void conv_f16(hipStream_t st, const _Float16* act, const _Float16* wgt, const float* bias,
                     _Float16* out, int B, int Hin, int Win, int Cin, int Cout,
                     int KH, int KW, int stride, int relu) {
  int Hout = (Hin + stride - 1) / stride;
  int Wout = (Win + stride - 1) / stride;
  int padT = (Hout - 1) * stride + KH - Hin; if (padT < 0) padT = 0;
  int pad = padT / 2;   // SAME lo-pad; all shapes here are square with KH==KW
  int M = B * Hout * Wout;
  dim3 g((M + BM - 1) / BM, (Cout + BN - 1) / BN);
  conv_implicit_gemm_wmma<<<g, dim3(256), 0, st>>>(act, wgt, bias, out, B, Hin, Win, Cin,
                                                   Hout, Wout, Cout, KH, KW, stride, pad, relu);
}

static void repack(hipStream_t st, const float* w, _Float16* out, int Cout, int Cin, int KH, int KW) {
  long n = (long)Cout * Cin * KH * KW;
  k_repack_w<<<eg(n), 256, 0, st>>>(w, out, Cout, Cin, KH, KW);
}

extern "C" void kernel_launch(void* const* d_in, const int* in_sizes, int n_in,
                              void* d_out, int out_size, void* d_ws, size_t ws_size,
                              hipStream_t stream) {
  (void)in_sizes; (void)n_in; (void)out_size; (void)ws_size;
  char* wsp = (char*)d_ws;
  auto a16 = [&](long elems) -> _Float16* {
    _Float16* r = (_Float16*)wsp; wsp += ((elems * 2 + 255) & ~(long)255); return r; };
  auto af = [&](long elems) -> float* {
    float* r = (float*)wsp; wsp += ((elems * 4 + 255) & ~(long)255); return r; };
  auto ai = [&](long elems) -> int* {
    int* r = (int*)wsp; wsp += ((elems * 4 + 255) & ~(long)255); return r; };

  // ---- inputs (setup_inputs dict order: x, then params in insertion order) ----
  const float* X = (const float*)d_in[0];
  auto F = [&](int i) { return (const float*)d_in[i]; };

  // ---- repack all conv weights to [Cout][k] f16 ----
  _Float16* w_stem = a16(64L * 147);        repack(stream, F(1), w_stem, 64, 3, 7, 7);
  _Float16* w_c3   = a16(512L * 576);       repack(stream, F(3), w_c3, 512, 64, 3, 3);
  _Float16* w_c4   = a16(1024L * 4608);     repack(stream, F(5), w_c4, 1024, 512, 3, 3);
  _Float16* w_c5   = a16(2048L * 9216);     repack(stream, F(7), w_c5, 2048, 1024, 3, 3);
  _Float16* w_l3   = a16(256L * 512);       repack(stream, F(9), w_l3, 256, 512, 1, 1);
  _Float16* w_l4   = a16(256L * 1024);      repack(stream, F(11), w_l4, 256, 1024, 1, 1);
  _Float16* w_l5   = a16(256L * 2048);      repack(stream, F(13), w_l5, 256, 2048, 1, 1);
  _Float16* w_s3   = a16(256L * 2304);      repack(stream, F(15), w_s3, 256, 256, 3, 3);
  _Float16* w_s4   = a16(256L * 2304);      repack(stream, F(17), w_s4, 256, 256, 3, 3);
  _Float16* w_s5   = a16(256L * 2304);      repack(stream, F(19), w_s5, 256, 256, 3, 3);
  _Float16* w_p6   = a16(256L * 18432);     repack(stream, F(21), w_p6, 256, 2048, 3, 3);
  _Float16* w_p7   = a16(256L * 2304);      repack(stream, F(23), w_p7, 256, 256, 3, 3);
  _Float16* w_ct[4]; _Float16* w_rt[4];
  const float* b_ct[4]; const float* b_rt[4];
  for (int t = 0; t < 4; t++) {
    w_ct[t] = a16(256L * 2304); repack(stream, F(25 + 2 * t), w_ct[t], 256, 256, 3, 3);
    b_ct[t] = F(26 + 2 * t);
    w_rt[t] = a16(256L * 2304); repack(stream, F(35 + 2 * t), w_rt[t], 256, 256, 3, 3);
    b_rt[t] = F(36 + 2 * t);
  }
  _Float16* w_co = a16(810L * 2304); repack(stream, F(33), w_co, 810, 256, 3, 3);
  _Float16* w_ro = a16(36L * 2304);  repack(stream, F(43), w_ro, 36, 256, 3, 3);

  // ---- activations ----
  _Float16* x16 = a16(2L * 512 * 512 * 3);
  k_nchw_to_nhwc_f16<<<eg(2L * 3 * 512 * 512), 256, 0, stream>>>(X, x16, 2, 3, 512, 512);

  _Float16* t_stem = a16(2L * 256 * 256 * 64);
  conv_f16(stream, x16, w_stem, F(2), t_stem, 2, 512, 512, 3, 64, 7, 7, 2, 1);
  _Float16* t_pool = a16(2L * 128 * 128 * 64);
  k_maxpool2<<<eg(2L * 128 * 128 * 64), 256, 0, stream>>>(t_stem, t_pool, 2, 256, 256, 64);

  _Float16* c3a = a16(2L * 64 * 64 * 512);
  conv_f16(stream, t_pool, w_c3, F(4), c3a, 2, 128, 128, 64, 512, 3, 3, 2, 1);
  _Float16* c4a = a16(2L * 32 * 32 * 1024);
  conv_f16(stream, c3a, w_c4, F(6), c4a, 2, 64, 64, 512, 1024, 3, 3, 2, 1);
  _Float16* c5a = a16(2L * 16 * 16 * 2048);
  conv_f16(stream, c4a, w_c5, F(8), c5a, 2, 32, 32, 1024, 2048, 3, 3, 2, 1);

  // FPN
  _Float16* p5l = a16(2L * 16 * 16 * 256);
  conv_f16(stream, c5a, w_l5, F(14), p5l, 2, 16, 16, 2048, 256, 1, 1, 1, 0);
  _Float16* p4l = a16(2L * 32 * 32 * 256);
  conv_f16(stream, c4a, w_l4, F(12), p4l, 2, 32, 32, 1024, 256, 1, 1, 1, 0);
  _Float16* p3l = a16(2L * 64 * 64 * 256);
  conv_f16(stream, c3a, w_l3, F(10), p3l, 2, 64, 64, 512, 256, 1, 1, 1, 0);
  _Float16* p4s = a16(2L * 32 * 32 * 256);
  k_add_up2<<<eg(2L * 32 * 32 * 256), 256, 0, stream>>>(p4l, p5l, p4s, 2, 32, 32, 256);
  _Float16* p3s = a16(2L * 64 * 64 * 256);
  k_add_up2<<<eg(2L * 64 * 64 * 256), 256, 0, stream>>>(p3l, p4s, p3s, 2, 64, 64, 256);
  _Float16* p3f = a16(2L * 64 * 64 * 256);
  conv_f16(stream, p3s, w_s3, F(16), p3f, 2, 64, 64, 256, 256, 3, 3, 1, 0);
  _Float16* p4f = a16(2L * 32 * 32 * 256);
  conv_f16(stream, p4s, w_s4, F(18), p4f, 2, 32, 32, 256, 256, 3, 3, 1, 0);
  _Float16* p5f = a16(2L * 16 * 16 * 256);
  conv_f16(stream, p5l, w_s5, F(20), p5f, 2, 16, 16, 256, 256, 3, 3, 1, 0);
  _Float16* p6f = a16(2L * 8 * 8 * 256);
  conv_f16(stream, c5a, w_p6, F(22), p6f, 2, 16, 16, 2048, 256, 3, 3, 2, 0);
  _Float16* p6r = a16(2L * 8 * 8 * 256);
  k_relu_f16<<<eg(2L * 8 * 8 * 256), 256, 0, stream>>>(p6f, p6r, 2L * 8 * 8 * 256);
  _Float16* p7f = a16(2L * 4 * 4 * 256);
  conv_f16(stream, p6r, w_p7, F(24), p7f, 2, 8, 8, 256, 256, 3, 3, 2, 0);

  // heads
  _Float16* ping = a16(2L * 64 * 64 * 256);
  _Float16* pong = a16(2L * 64 * 64 * 256);
  _Float16* cls_raw = a16(2L * 64 * 64 * 810);
  _Float16* reg_raw = a16(2L * 64 * 64 * 36);
  float* classify = af(2L * NTOT * 90);
  float* boxes = af(2L * NTOT * 4);

  const _Float16* feats[5] = {p3f, p4f, p5f, p6f, p7f};
  const int HWs[5] = {64, 32, 16, 8, 4};
  const float strides_[5] = {8.0f, 16.0f, 32.0f, 64.0f, 128.0f};
  long base = 0;
  for (int l = 0; l < 5; l++) {
    int H = HWs[l];
    // classification head
    const _Float16* cur = feats[l];
    for (int t = 0; t < 4; t++) {
      _Float16* dst = (t & 1) ? pong : ping;
      conv_f16(stream, cur, w_ct[t], b_ct[t], dst, 2, H, H, 256, 256, 3, 3, 1, 1);
      cur = dst;
    }
    conv_f16(stream, cur, w_co, F(34), cls_raw, 2, H, H, 256, 810, 3, 3, 1, 0);
    k_sigmoid_store<<<eg(2L * H * H * 810), 256, 0, stream>>>(
        cls_raw, classify, 2, (long)H * H * 810, (long)NTOT * 90, base * 90);
    // regression head
    cur = feats[l];
    for (int t = 0; t < 4; t++) {
      _Float16* dst = (t & 1) ? pong : ping;
      conv_f16(stream, cur, w_rt[t], b_rt[t], dst, 2, H, H, 256, 256, 3, 3, 1, 1);
      cur = dst;
    }
    conv_f16(stream, cur, w_ro, F(44), reg_raw, 2, H, H, 256, 36, 3, 3, 1, 0);
    k_decode<<<eg(2L * H * H * 9), 256, 0, stream>>>(reg_raw, boxes, 2, H, H,
                                                     strides_[l], NTOT, base);
    base += (long)H * H * 9;
  }

  // post-processing
  float* maxs = af(2L * NTOT);
  int* lab = ai(2L * NTOT);
  float* swork = af(2L * NTOT);
  k_maxclass<<<eg(2L * NTOT), 256, 0, stream>>>(classify, maxs, lab, 2, NTOT);

  float* outB = (float*)d_out;
  float* outS = outB + 2L * POST * 4;
  float* outL = outS + 2L * POST;
  k_nms<<<2, 256, 0, stream>>>(maxs, lab, boxes, swork, outB, outS, outL, NTOT);
}
